// LatentNet_61899068670079
// MI455X (gfx1250) — matrix operand order repaired and checked by reading, not compiled, not run on hardware
//
#include <hip/hip_runtime.h>

typedef __attribute__((ext_vector_type(16))) _Float16 v16h;
typedef __attribute__((ext_vector_type(8)))  _Float16 v8h;
typedef __attribute__((ext_vector_type(8)))  float    v8f;

#define B_DIM 512
#define T_DIM 1024
#define IN_DIM 6
#define N_DIM 64
// alpha = 0.2, sigma_rec = 0.15, scale = sqrt(2*alpha*sigma^2) = sqrt(0.009)
#define NOISE_SCALE 0.094868329805051374f
#define DECAY 0.8f
#define ALPHA_C 0.2f

// LDS tile: 16 rows (batch) x 104 f16 cols (64 state | 6 u | zero pad to K=96;
// row stride 104 halves = 208 B = 52 dwords -> 16 rows land on distinct
// 4-dword bank groups => conflict-free ds_load_b128 A-fragment reads).
#define LROW 104

__global__ __launch_bounds__(32)
void rnn_scan_wmma(const float* __restrict__ u,
                   const float* __restrict__ noise,
                   const float* __restrict__ W_rec,
                   const float* __restrict__ W_in,
                   float* __restrict__ out)
{
    __shared__ __align__(16) _Float16 lds[16][LROW];

    const int lane = threadIdx.x;     // 0..31
    const int half = lane >> 4;       // 0 or 1
    const int m    = lane & 15;
    const int b0   = blockIdx.x * 16; // batch tile base

    // ---- zero the LDS tile (incl. zero-pad K columns 70..95) ----
    {
        unsigned* p = (unsigned*)&lds[0][0];
        for (int i = lane; i < (16 * LROW) / 2; i += 32) p[i] = 0u;
    }

    // ---- build time-invariant B fragments: Bext[k,j], K=96, N=64 ----
    // Bext[k,j] = W_rec[j,k] (k<64) | W_in[j,k-64] (64<=k<70) | 0
    // B v16h layout: elem h <-> K = 32c + 8*half + h (h<8) | 32c + 16 + 8*half + (h-8) (h>=8), N = 16*nt + m
    v16h bfrag[3][4];
    #pragma unroll
    for (int c = 0; c < 2; ++c) {              // chunks 0,1: pure W_rec, no branches
        #pragma unroll
        for (int nt = 0; nt < 4; ++nt) {
            const int j = nt * 16 + m;
            const float* wr = W_rec + j * 64 + 32 * c + 8 * half;
            v16h f;
            #pragma unroll
            for (int h = 0; h < 16; ++h) f[h] = (_Float16)wr[h < 8 ? h : h + 8];
            bfrag[c][nt] = f;
        }
    }
    #pragma unroll
    for (int nt = 0; nt < 4; ++nt) {           // chunk 2: W_in rows 64..69, rest zero
        const int j = nt * 16 + m;
        v16h f = {};
        if (half == 0) {                       // K = 64 + h, nonzero only for h < 6
            #pragma unroll
            for (int h = 0; h < 6; ++h) f[h] = (_Float16)W_in[j * 6 + h];
        }
        bfrag[2][nt] = f;
    }

    // ---- state fragments (C/D layout: elem r <-> row M = r + 8*half, col j = 16*nt + m) ----
    v8f s[4];
    #pragma unroll
    for (int nt = 0; nt < 4; ++nt) { v8f z = {}; s[nt] = z; }

    // ---- out[:, 0, :] = 0 ----
    {
        float* op0 = out + ((size_t)(b0 + 8 * half) * T_DIM) * N_DIM + m;
        #pragma unroll
        for (int nt = 0; nt < 4; ++nt)
            #pragma unroll
            for (int r = 0; r < 8; ++r)
                op0[(size_t)r * T_DIM * N_DIM + nt * 16] = 0.0f;
    }

    // ---- u staging: 96 elements (16 rows x 6), 3 per lane ----
    const float* up[3];
    _Float16*    ulp[3];
    #pragma unroll
    for (int k = 0; k < 3; ++k) {
        const int e   = lane + 32 * k;   // 0..95
        const int row = e / 6;
        const int col = e % 6;
        up[k]  = u + ((size_t)(b0 + row) * T_DIM) * IN_DIM + col; // t = 0
        ulp[k] = &lds[row][64 + col];
    }
    #pragma unroll
    for (int k = 0; k < 3; ++k) { *ulp[k] = (_Float16)(*up[k]); up[k] += IN_DIM; }

    // ---- per-lane global pointers (advance by N per step; strided offsets fit 24-bit imm) ----
    const float* np = noise + ((size_t)(b0 + 8 * half) * T_DIM) * N_DIM + m;      // t = 0
    float*       op = out   + ((size_t)(b0 + 8 * half) * T_DIM + 1) * N_DIM + m;  // t = 1

    // A-fragment base address in LDS for this lane
    const _Float16* arow = &lds[m][8 * half];

    // ---- noise prefetch for t = 0 ----
    float cn[4][8];
    #pragma unroll
    for (int nt = 0; nt < 4; ++nt)
        #pragma unroll
        for (int r = 0; r < 8; ++r)
            cn[nt][r] = np[(size_t)r * T_DIM * N_DIM + nt * 16];
    np += N_DIM;

    // ================= main recurrence: 1023 dependent steps =================
    #pragma unroll 1
    for (int t = 0; t < T_DIM - 1; ++t) {
        // All six A-fragment loads issue up front (distinct regs per chunk ->
        // progressive dscnt waits instead of full serialization per chunk).
        const v8h lo0 = *(const v8h*)(arow +  0);
        const v8h hi0 = *(const v8h*)(arow + 16);
        const v8h lo1 = *(const v8h*)(arow + 32);
        const v8h hi1 = *(const v8h*)(arow + 48);
        const v8h lo2 = *(const v8h*)(arow + 64);
        const v8h hi2 = *(const v8h*)(arow + 80);
        const v16h a0 = __builtin_shufflevector(lo0, hi0, 0,1,2,3,4,5,6,7,8,9,10,11,12,13,14,15);
        const v16h a1 = __builtin_shufflevector(lo1, hi1, 0,1,2,3,4,5,6,7,8,9,10,11,12,13,14,15);
        const v16h a2 = __builtin_shufflevector(lo2, hi2, 0,1,2,3,4,5,6,7,8,9,10,11,12,13,14,15);

        // C init = sigma * noise_t
        v8f acc[4];
        #pragma unroll
        for (int nt = 0; nt < 4; ++nt)
            #pragma unroll
            for (int r = 0; r < 8; ++r)
                acc[nt][r] = NOISE_SCALE * cn[nt][r];

        // prefetch noise for t+1 (last iter reads t=1023: in-bounds, unused)
        #pragma unroll
        for (int nt = 0; nt < 4; ++nt)
            #pragma unroll
            for (int r = 0; r < 8; ++r)
                cn[nt][r] = np[(size_t)r * T_DIM * N_DIM + nt * 16];
        np += N_DIM;

        // acc += [s_t | u_t | 0] @ Bext   (3 K-chunks x 4 N-tiles = 12 WMMA)
        #pragma unroll
        for (int nt = 0; nt < 4; ++nt)
            acc[nt] = __builtin_amdgcn_wmma_f32_16x16x32_f16(
                false, a0, false, bfrag[0][nt], (short)0, acc[nt], false, false);
        #pragma unroll
        for (int nt = 0; nt < 4; ++nt)
            acc[nt] = __builtin_amdgcn_wmma_f32_16x16x32_f16(
                false, a1, false, bfrag[1][nt], (short)0, acc[nt], false, false);
        #pragma unroll
        for (int nt = 0; nt < 4; ++nt)
            acc[nt] = __builtin_amdgcn_wmma_f32_16x16x32_f16(
                false, a2, false, bfrag[2][nt], (short)0, acc[nt], false, false);

        // s_{t+1} = 0.8*s_t + 0.2*relu(pre); write out + LDS (f16) for next step's A
        #pragma unroll
        for (int nt = 0; nt < 4; ++nt) {
            v8f sn;
            #pragma unroll
            for (int r = 0; r < 8; ++r)
                sn[r] = DECAY * s[nt][r] + ALPHA_C * fmaxf(acc[nt][r], 0.0f);
            s[nt] = sn;
            #pragma unroll
            for (int r = 0; r < 8; ++r) {
                op[(size_t)r * T_DIM * N_DIM + nt * 16] = sn[r];
                lds[r + 8 * half][nt * 16 + m] = (_Float16)sn[r];
            }
        }
        op += N_DIM;

        // stage u_{t+1} (last iter stages u[1023]: in-bounds, unused)
        #pragma unroll
        for (int k = 0; k < 3; ++k) { *ulp[k] = (_Float16)(*up[k]); up[k] += IN_DIM; }
    }
}

extern "C" void kernel_launch(void* const* d_in, const int* in_sizes, int n_in,
                              void* d_out, int out_size, void* d_ws, size_t ws_size,
                              hipStream_t stream) {
    const float* u     = (const float*)d_in[0];
    const float* noise = (const float*)d_in[1];
    const float* W_rec = (const float*)d_in[2];
    const float* W_in  = (const float*)d_in[3];
    float*       out   = (float*)d_out;

    dim3 grid(B_DIM / 16);  // 32 workgroups, one wave each
    dim3 block(32);
    hipLaunchKernelGGL(rnn_scan_wmma, grid, block, 0, stream,
                       u, noise, W_rec, W_in, out);
}
